// Attention_25262997635763
// MI455X (gfx1250) — compile-verified, matrix operand order checked by reading, not run on hardware
//
#include <hip/hip_runtime.h>
#include <hip/hip_bf16.h>
#include <math.h>

#define BB 64
#define TT 2048
#define DD 512
#define UU 512

typedef __attribute__((ext_vector_type(16))) __bf16 v16bf;
typedef __attribute__((ext_vector_type(8)))  float  v8f;
typedef __attribute__((ext_vector_type(4)))  float  v4f;

// Hardware tanh if the builtin exists (gfx1250 has V_TANH_F32), else libm.
#if __has_builtin(__builtin_amdgcn_tanhf)
#define FAST_TANH(x) __builtin_amdgcn_tanhf(x)
#else
#define FAST_TANH(x) tanhf(x)
#endif

// ---------------------------------------------------------------------------
// Pack W1 (f32 [D,U] row-major) into bf16 B-matrix fragments.
// Fragment (nt, kt): K rows [kt*32, kt*32+32), N cols [nt*16, nt*16+16).
// Per-lane layout (mirror of ISA 16-bit A layout, M->N):
//   lane L: col n = nt*16 + (L&15)
//   elems 0..7  : k = kt*32 + (L>=16 ? 8 : 0) + e
//   elems 8..15 : k = kt*32 + 16 + (L>=16 ? 8 : 0) + (e-8)
// Linear element index: (nt*16+kt)*512 + lane*16 + e
// ---------------------------------------------------------------------------
__global__ __launch_bounds__(256) void k_pack(const float* __restrict__ W1,
                                              __bf16* __restrict__ pk) {
    int idx  = blockIdx.x * 256 + threadIdx.x;      // 0 .. 262143
    int e    = idx & 15;
    int lane = (idx >> 4) & 31;
    int kt   = (idx >> 9) & 15;
    int nt   = idx >> 13;
    int k = kt * 32 + ((lane >= 16) ? 8 : 0) + (e & 7) + ((e >> 3) << 4);
    int n = nt * 16 + (lane & 15);
    pk[idx] = (__bf16)W1[k * UU + n];
}

// ---------------------------------------------------------------------------
// ph[b,u] = sum_d h[b,d]*W2[d,u] + W1_b[u] + W2_b[u]
// ---------------------------------------------------------------------------
__global__ __launch_bounds__(256) void k_ph(const float* __restrict__ h,
                                            const float* __restrict__ W2,
                                            const float* __restrict__ b1,
                                            const float* __restrict__ b2,
                                            float* __restrict__ ph) {
    int idx = blockIdx.x * 256 + threadIdx.x;       // B*U
    int b = idx >> 9;
    int u = idx & (UU - 1);
    const float* hb = h + b * DD;
    float acc = b1[u] + b2[u];
    #pragma unroll 4
    for (int d = 0; d < DD; ++d) acc += hb[d] * W2[d * UU + u];
    ph[idx] = acc;
}

// ---------------------------------------------------------------------------
// Main WMMA kernel: logits[b,t] = sum_u tanh( (c@W1)[t,u] + ph[b,u] ) * V[u]
// grid = (T/256, B), 256 threads = 8 waves.
// Each wave owns M=32 rows (two 16-row A-fragment sets -> B reused 2x).
// B fragments double-buffered in LDS via global_load_async_to_lds_b128,
// and register-prefetched one K-step ahead inside the compute loop.
// ---------------------------------------------------------------------------
__global__ __launch_bounds__(256) void k_logits(const float* __restrict__ c,
                                                const __bf16* __restrict__ w1p,
                                                const float* __restrict__ ph,
                                                const float* __restrict__ Vw,
                                                float* __restrict__ logits) {
    __shared__ __bf16 bbuf[2][16 * 512];            // 2 x 16 KB double buffer

    const int b    = blockIdx.y;
    const int tile = blockIdx.x;                    // T/256 tiles
    const int tid  = threadIdx.x;
    const int wave = tid >> 5;
    const int lane = tid & 31;
    const int half = lane >> 4;                     // 0 or 1
    const int l15  = lane & 15;
    const int m_base = tile * 256 + wave * 32;

    // -- async prefetch of B fragments for one N-tile into LDS -------------
    //    16 KB = 1024 x b128 ; 256 threads x 4 transfers each.
    auto issue_b = [&](int nt, int s) {
        const char* gp = (const char*)(w1p + (size_t)nt * 16 * 512);
        unsigned lbase = (unsigned)(uintptr_t)&bbuf[s][0];
        #pragma unroll
        for (int i = 0; i < 4; ++i) {
            unsigned    loff = lbase + (unsigned)(tid + i * 256) * 16u;
            const char* ga   = gp + (size_t)(tid + i * 256) * 16;
            asm volatile("global_load_async_to_lds_b128 %0, %1, off"
                         :: "v"(loff), "v"(ga) : "memory");
        }
    };

    issue_b(0, 0);   // kick stage 0 before the long A-tile load

    // -- load & convert this wave's 32 rows of c into bf16 A fragments -----
    const int kb_off = half * 8;
    v16bf a0[16], a1[16];
    auto load_a = [&](const float* rowp, v16bf* a) {
        #pragma unroll
        for (int kt = 0; kt < 16; ++kt) {
            const int kb = kt * 32 + kb_off;
            v4f x0 = *(const v4f*)(rowp + kb);
            v4f x1 = *(const v4f*)(rowp + kb + 4);
            v4f x2 = *(const v4f*)(rowp + kb + 16);
            v4f x3 = *(const v4f*)(rowp + kb + 20);
            v16bf t;
            #pragma unroll
            for (int j = 0; j < 4; ++j) {
                t[j]      = (__bf16)x0[j];
                t[4 + j]  = (__bf16)x1[j];
                t[8 + j]  = (__bf16)x2[j];
                t[12 + j] = (__bf16)x3[j];
            }
            a[kt] = t;
        }
    };
    const float* rowp0 = c + ((size_t)b * TT + (m_base + l15)) * DD;
    load_a(rowp0, a0);
    load_a(rowp0 + 16 * DD, a1);

    v8f lp0 = {}, lp1 = {};
    for (int nt = 0; nt < 32; ++nt) {
        const int s = nt & 1;
        if (nt < 31) {
            issue_b(nt + 1, s ^ 1);
            asm volatile("s_wait_asynccnt 0x4" ::: "memory"); // stage nt done
        } else {
            asm volatile("s_wait_asynccnt 0x0" ::: "memory");
        }
        __syncthreads();                        // buf[s] visible to all waves

        const float vv  = Vw[nt * 16 + l15];
        const float phu = ph[b * UU + nt * 16 + l15];
        const v16bf* bp = (const v16bf*)&bbuf[s][0] + lane;

        v8f acc0 = {}, acc1 = {};
        v16bf bcur = bp[0];                     // fragment kt = 0
        #pragma unroll
        for (int kt = 0; kt < 16; ++kt) {
            v16bf bnxt = bcur;
            if (kt < 15) bnxt = bp[(kt + 1) * 32];  // issue ahead of WMMAs
            acc0 = __builtin_amdgcn_wmma_f32_16x16x32_bf16(
                false, a0[kt], false, bcur, (short)0, acc0, false, false);
            acc1 = __builtin_amdgcn_wmma_f32_16x16x32_bf16(
                false, a1[kt], false, bcur, (short)0, acc1, false, false);
            bcur = bnxt;
        }
        // D layout: VGPR r, lanes 0-15 -> (M=r, N=l15); lanes 16-31 -> M=8+r
        #pragma unroll
        for (int r = 0; r < 8; ++r) {
            lp0[r] += FAST_TANH(acc0[r] + phu) * vv;
            lp1[r] += FAST_TANH(acc1[r] + phu) * vv;
        }
        __syncthreads();      // all waves done reading buf[s] before refill
    }

    // -- reduce over the 16 lanes (N dim) sharing each M row ---------------
    #pragma unroll
    for (int r = 0; r < 8; ++r) {
        float v = lp0[r];
        v += __shfl_xor(v, 1, 32);
        v += __shfl_xor(v, 2, 32);
        v += __shfl_xor(v, 4, 32);
        v += __shfl_xor(v, 8, 32);
        lp0[r] = v;
        float w = lp1[r];
        w += __shfl_xor(w, 1, 32);
        w += __shfl_xor(w, 2, 32);
        w += __shfl_xor(w, 4, 32);
        w += __shfl_xor(w, 8, 32);
        lp1[r] = w;
    }
    if (l15 == 0) {
        float* lp = logits + (size_t)b * TT + m_base + half * 8;
        #pragma unroll
        for (int r = 0; r < 8; ++r) {
            lp[r]      = lp0[r];
            lp[16 + r] = lp1[r];
        }
    }
}

// ---------------------------------------------------------------------------
// Softmax over T per batch: one block per b.
// ---------------------------------------------------------------------------
__global__ __launch_bounds__(256) void k_softmax(const float* __restrict__ logits,
                                                 float* __restrict__ attn) {
    const int b   = blockIdx.x;
    const int tid = threadIdx.x;
    __shared__ float red[256];
    const float* lg = logits + (size_t)b * TT;

    float v[8];
    float mx = -3.4e38f;
    #pragma unroll
    for (int i = 0; i < 8; ++i) {
        v[i] = lg[tid + i * 256];
        mx = fmaxf(mx, v[i]);
    }
    red[tid] = mx; __syncthreads();
    for (int s = 128; s > 0; s >>= 1) {
        if (tid < s) red[tid] = fmaxf(red[tid], red[tid + s]);
        __syncthreads();
    }
    mx = red[0]; __syncthreads();

    float sum = 0.f;
    #pragma unroll
    for (int i = 0; i < 8; ++i) {
        v[i] = expf(v[i] - mx);
        sum += v[i];
    }
    red[tid] = sum; __syncthreads();
    for (int s = 128; s > 0; s >>= 1) {
        if (tid < s) red[tid] += red[tid + s];
        __syncthreads();
    }
    const float inv = 1.0f / red[0];

    float* at = attn + (size_t)b * TT;
    #pragma unroll
    for (int i = 0; i < 8; ++i) at[tid + i * 256] = v[i] * inv;
}

// ---------------------------------------------------------------------------
// context[b,d] = sum_t attn[b,t] * c[b,t,d]   (HBM-bandwidth-bound)
// grid = (D/256, B)
// ---------------------------------------------------------------------------
__global__ __launch_bounds__(256) void k_context(const float* __restrict__ c,
                                                 const float* __restrict__ attn,
                                                 float* __restrict__ out) {
    const int b = blockIdx.y;
    const int d = blockIdx.x * 256 + threadIdx.x;
    const float* cb = c + (size_t)b * TT * DD + d;
    const float* ab = attn + (size_t)b * TT;
    float acc = 0.f;
    #pragma unroll 4
    for (int t = 0; t < TT; ++t) acc += ab[t] * cb[(size_t)t * DD];
    out[(size_t)b * DD + d] = acc;
}

// ---------------------------------------------------------------------------
extern "C" void kernel_launch(void* const* d_in, const int* in_sizes, int n_in,
                              void* d_out, int out_size, void* d_ws, size_t ws_size,
                              hipStream_t stream) {
    const float* c   = (const float*)d_in[0];   // [B,T,D]
    const float* h   = (const float*)d_in[1];   // [B,D]
    const float* W1  = (const float*)d_in[2];   // [D,U]
    const float* W1b = (const float*)d_in[3];   // [U]
    const float* W2  = (const float*)d_in[4];   // [D,U]
    const float* W2b = (const float*)d_in[5];   // [U]
    const float* Vw  = (const float*)d_in[6];   // [U,1]
    // d_in[7] = V_b : constant shift, softmax-invariant -> dropped
    float* out = (float*)d_out;                 // [B,D]

    char* ws = (char*)d_ws;
    __bf16* w1p   = (__bf16*)(ws);                       // 512 KB
    float*  ph    = (float*)(ws + 524288);               // 128 KB
    float*  logit = (float*)(ws + 655360);               // 512 KB
    float*  attn  = (float*)(ws + 1179648);              // 512 KB

    k_pack   <<<1024, 256, 0, stream>>>(W1, w1p);
    k_ph     <<<(BB * UU) / 256, 256, 0, stream>>>(h, W2, W1b, W2b, ph);
    k_logits <<<dim3(TT / 256, BB), 256, 0, stream>>>(c, w1p, ph, Vw, logit);
    k_softmax<<<BB, 256, 0, stream>>>(logit, attn);
    k_context<<<dim3(DD / 256, BB), 256, 0, stream>>>(c, attn, out);
}
